// AttendAndSpell_54202487275965
// MI455X (gfx1250) — compile-verified, matrix operand order checked by reading, not run on hardware
//
#include <hip/hip_runtime.h>

// ---------------------------------------------------------------------------
// AttendAndSpell decoder for MI455X (gfx1250, wave32, WMMA).
// All GEMMs use v_wmma_f32_16x16x32_f16 (f16 inputs, fp32 accumulate).
// The K-dimension of all f16 GEMM operands is stored "fragment-permuted"
// (within each 32-block: [0..7,16..23,8..15,24..31]) so each lane's WMMA
// fragment is one contiguous 32B region -> two adjacent global_load_b128
// directly into the WMMA source VGPRs, no register packing moves.
// One-hot @ Wih is folded into a row gather. Everything stays L2-resident.
// ---------------------------------------------------------------------------

#define HID   512
#define VOCAB 10000
#define BATCH 32
#define RLEN  1024
#define TLEN  256
#define LN_EPS 1e-5f

typedef __attribute__((ext_vector_type(16))) _Float16 v16h;
typedef __attribute__((ext_vector_type(8)))  float    v8f;

__device__ __forceinline__ float sigmoidf_(float x) { return 1.0f / (1.0f + __expf(-x)); }

// K-index permutation: within each 32-block swap groups 1 and 2 (of 8).
// Valid only when the base offset of the K-range is a multiple of 32.
__device__ __forceinline__ long frag_perm(long i) {
    long g = (i >> 3) & 3;
    g = (g == 1) ? 2 : (g == 2) ? 1 : g;
    return (i & ~24L) | (g << 3);
}

// ---------------------------------------------------------------------------
// f32 -> f16 convert with fragment permutation (row stride multiple of 32).
// ---------------------------------------------------------------------------
__global__ void cvt_f16_perm_kernel(const float* __restrict__ src, _Float16* __restrict__ dst, long n) {
    long i = (long)blockIdx.x * blockDim.x + threadIdx.x;
    long stride = (long)gridDim.x * blockDim.x;
    for (; i < n; i += stride) dst[frag_perm(i)] = (_Float16)src[i];
}

// f32 [K,N] row-major -> f16 transposed+permuted dst[n*dstStride + dstOff + perm(k)]
__global__ void transpose_cvt_kernel(const float* __restrict__ src, _Float16* __restrict__ dst,
                                     int K, int N, int dstStride, int dstOff) {
    long i = (long)blockIdx.x * blockDim.x + threadIdx.x;
    long total = (long)K * N;
    if (i >= total) return;
    int k = (int)(i / N);
    int n = (int)(i % N);
    dst[(long)n * dstStride + dstOff + frag_perm(k)] = (_Float16)src[i];
}

// ---------------------------------------------------------------------------
// WMMA GEMM: C[M,N] (fp32) = A[M,K] (f16, K fragment-permuted) * B (f16 stored
// transposed+permuted as BT[N,K]). M,K multiples of 32; N multiple of 16.
// Block = 256 threads = 8 waves: 2 M-tiles x 4 N-tile-pairs -> 32x128/block.
// Each wave: 1 A-frag reused for 2 B-frags (2 WMMAs per 3x32B loads/K-step).
// Thanks to the permuted layout, lane L's fragment = 32 contiguous bytes at
// row(L&15) + (L>>4)*16 halves: loads go straight into WMMA source VGPRs.
// ---------------------------------------------------------------------------
__global__ void gemm_f16_wmma_kernel(const _Float16* __restrict__ A, int lda,
                                     const _Float16* __restrict__ BT, int ldb,
                                     float* __restrict__ C, int ldc,
                                     int N, int K) {
    const int lane = threadIdx.x & 31;
    const int wave = threadIdx.x >> 5;
    const int m0 = blockIdx.y * 32 + (wave >> 2) * 16;
    const int n0 = (blockIdx.x * 8 + (wave & 3) * 2) * 16;
    const int n1 = n0 + 16;
    if (n0 >= N) return;              // uniform within wave
    const bool has2 = (n1 < N);

    const int sel = (lane >> 4) * 16; // which 16-half group of the 32-block
    const _Float16* aptr = A  + (long)(m0 + (lane & 15)) * lda + sel;
    const _Float16* b0p  = BT + (long)(n0 + (lane & 15)) * ldb + sel;
    const _Float16* b1p  = has2 ? BT + (long)(n1 + (lane & 15)) * ldb + sel : b0p;

    v8f acc0 = {0.f, 0.f, 0.f, 0.f, 0.f, 0.f, 0.f, 0.f};
    v8f acc1 = {0.f, 0.f, 0.f, 0.f, 0.f, 0.f, 0.f, 0.f};

#pragma unroll 4
    for (int k = 0; k < K; k += 32) {
        v16h av  = *reinterpret_cast<const v16h*>(aptr + k);
        v16h b0v = *reinterpret_cast<const v16h*>(b0p + k);
        v16h b1v = *reinterpret_cast<const v16h*>(b1p + k);
        __builtin_prefetch(b0p + k + 256, 0, 1);
        acc0 = __builtin_amdgcn_wmma_f32_16x16x32_f16(false, av, false, b0v,
                                                      (short)0, acc0, false, false);
        acc1 = __builtin_amdgcn_wmma_f32_16x16x32_f16(false, av, false, b1v,
                                                      (short)0, acc1, false, false);
    }

    const int crow0 = m0 + (lane >> 4) * 8;
    const int ccol  = n0 + (lane & 15);
#pragma unroll
    for (int j = 0; j < 8; ++j)
        C[(long)(crow0 + j) * ldc + ccol] = acc0[j];
    if (has2) {
#pragma unroll
        for (int j = 0; j < 8; ++j)
            C[(long)(crow0 + j) * ldc + ccol + 16] = acc1[j];
    }
}

// ---------------------------------------------------------------------------
// Fused epilogue: v = X[row]+bias ; (relu) ; (+= resid[row]) ; LayerNorm(D)
// f16 output is written fragment-permuted (it feeds a GEMM A operand or the attention kernels;
// only used with D multiple of 32). f32 output stays in natural order.
// ---------------------------------------------------------------------------
__global__ void bias_act_ln_kernel(const float* __restrict__ X,
                                   const float* __restrict__ bias,
                                   const float* __restrict__ resid,
                                   const float* __restrict__ gamma,
                                   const float* __restrict__ beta,
                                   _Float16* __restrict__ out16, long stride16,
                                   float* __restrict__ out32,   long stride32,
                                   int D, int doRelu) {
    __shared__ float red[2][8];
    const long row = blockIdx.x;
    const float* x = X + row * D;
    const float* r = resid ? resid + row * D : nullptr;

    float s = 0.f, s2 = 0.f;
    for (int i = threadIdx.x; i < D; i += blockDim.x) {
        float v = x[i] + bias[i];
        if (doRelu) v = fmaxf(v, 0.f);
        if (r) v += r[i];
        s += v; s2 += v * v;
    }
#pragma unroll
    for (int o = 16; o > 0; o >>= 1) { s += __shfl_xor(s, o, 32); s2 += __shfl_xor(s2, o, 32); }
    const int w = threadIdx.x >> 5, lane = threadIdx.x & 31;
    if (lane == 0) { red[0][w] = s; red[1][w] = s2; }
    __syncthreads();
    if (w == 0) {
        s  = (lane < 8) ? red[0][lane] : 0.f;
        s2 = (lane < 8) ? red[1][lane] : 0.f;
#pragma unroll
        for (int o = 4; o > 0; o >>= 1) { s += __shfl_xor(s, o, 32); s2 += __shfl_xor(s2, o, 32); }
        if (lane == 0) { red[0][0] = s; red[1][0] = s2; }
    }
    __syncthreads();
    const float mean = red[0][0] / D;
    const float var  = red[1][0] / D - mean * mean;
    const float rstd = rsqrtf(var + LN_EPS);

    for (int i = threadIdx.x; i < D; i += blockDim.x) {
        float v = x[i] + bias[i];
        if (doRelu) v = fmaxf(v, 0.f);
        if (r) v += r[i];
        float y = (v - mean) * rstd * gamma[i] + beta[i];
        if (out16) out16[row * stride16 + frag_perm(i)] = (_Float16)y;
        if (out32) out32[row * stride32 + i] = y;
    }
}

// ---------------------------------------------------------------------------
// Attention scores: e[b,r] = dot(s_phi[b,:], h_psi[b,r,:]) over 512 (f16 in).
// Both operands carry the same K-permutation -> dot is permutation-invariant.
// One wave per (b,r); 16 halves per lane, shfl tree-reduce.
// ---------------------------------------------------------------------------
struct alignas(16) Half8 { _Float16 h[8]; };

__global__ void attn_e_kernel(const _Float16* __restrict__ sphi,
                              const _Float16* __restrict__ hpsi,
                              float* __restrict__ e) {
    const int lane = threadIdx.x & 31;
    const int wid  = (blockIdx.x * blockDim.x + threadIdx.x) >> 5;
    const int b = wid >> 10;
    const int r = wid & (RLEN - 1);
    const _Float16* sp = sphi + (long)b * HID + lane * 16;
    const _Float16* hp = hpsi + ((long)b * RLEN + r) * HID + lane * 16;
    Half8 s0 = *reinterpret_cast<const Half8*>(sp);
    Half8 s1 = *reinterpret_cast<const Half8*>(sp + 8);
    Half8 h0 = *reinterpret_cast<const Half8*>(hp);
    Half8 h1 = *reinterpret_cast<const Half8*>(hp + 8);
    float acc = 0.f;
#pragma unroll
    for (int i = 0; i < 8; ++i) {
        acc += (float)s0.h[i] * (float)h0.h[i];
        acc += (float)s1.h[i] * (float)h1.h[i];
    }
#pragma unroll
    for (int o = 16; o > 0; o >>= 1) acc += __shfl_xor(acc, o, 32);
    if (lane == 0) e[(long)b * RLEN + r] = acc;
}

// softmax over R per row; block per row, 256 threads.
__global__ void softmax_rows_kernel(const float* __restrict__ e, float* __restrict__ alpha, int R) {
    __shared__ float red[8];
    const long row = blockIdx.x;
    const float* x = e + row * R;
    const int w = threadIdx.x >> 5, lane = threadIdx.x & 31;

    float m = -1e30f;
    for (int i = threadIdx.x; i < R; i += blockDim.x) m = fmaxf(m, x[i]);
#pragma unroll
    for (int o = 16; o > 0; o >>= 1) m = fmaxf(m, __shfl_xor(m, o, 32));
    if (lane == 0) red[w] = m;
    __syncthreads();
    if (w == 0) {
        m = (lane < 8) ? red[lane] : -1e30f;
#pragma unroll
        for (int o = 4; o > 0; o >>= 1) m = fmaxf(m, __shfl_xor(m, o, 32));
        if (lane == 0) red[0] = m;
    }
    __syncthreads();
    m = red[0];
    __syncthreads();

    float s = 0.f;
    for (int i = threadIdx.x; i < R; i += blockDim.x) s += __expf(x[i] - m);
#pragma unroll
    for (int o = 16; o > 0; o >>= 1) s += __shfl_xor(s, o, 32);
    if (lane == 0) red[w] = s;
    __syncthreads();
    if (w == 0) {
        s = (lane < 8) ? red[lane] : 0.f;
#pragma unroll
        for (int o = 4; o > 0; o >>= 1) s += __shfl_xor(s, o, 32);
        if (lane == 0) red[0] = s;
    }
    __syncthreads();
    const float inv = 1.f / red[0];
    for (int i = threadIdx.x; i < R; i += blockDim.x)
        alpha[row * R + i] = __expf(x[i] - m) * inv;
}

// context: c[b,p] = sum_r alpha[b,r] * h_psi[b,r,p] ; reads and writes the
// SAME (permuted) position p -> permutation-transparent. Writes xA0[b, 0:512].
__global__ void attn_c_kernel(const float* __restrict__ alpha,
                              const _Float16* __restrict__ hpsi,
                              _Float16* __restrict__ xA0) {
    const int b = blockIdx.x;
    const int d = threadIdx.x;  // 256 threads, 2 cols each
    const _Float16* hp = hpsi + (long)b * RLEN * HID;
    const float* al = alpha + (long)b * RLEN;
    float a0 = 0.f, a1 = 0.f;
    for (int r = 0; r < RLEN; ++r) {
        float a = al[r];
        a0 += a * (float)hp[(long)r * HID + d];
        a1 += a * (float)hp[(long)r * HID + 256 + d];
    }
    xA0[(long)b * 1024 + d]       = (_Float16)a0;
    xA0[(long)b * 1024 + 256 + d] = (_Float16)a1;
}

// ---------------------------------------------------------------------------
// LSTM cell pointwise: gates [32,2048] fp32 from GEMM (natural order); add
// bih+bhh and (optionally) the one-hot-gathered Wih row; i,f,g,o split;
// update c; emit h as f16 (fragment-permuted, feeds GEMM A) into one or two
// destinations. grid 32*512/256 threads.
// ---------------------------------------------------------------------------
__global__ void lstm_cell_kernel(const float* __restrict__ gates,
                                 const float* __restrict__ bih,
                                 const float* __restrict__ bhh,
                                 const float* __restrict__ embW,  // nullable: Wih [*,2048] fp32
                                 const int* __restrict__ y, int t, int T,
                                 float* __restrict__ cst,         // [32,512] in/out
                                 _Float16* __restrict__ outA, int strideA,
                                 _Float16* __restrict__ outB, int strideB) {
    const int g = blockIdx.x * blockDim.x + threadIdx.x;
    const int b = g >> 9;
    const int d = g & 511;
    const float* gr = gates + (long)b * 2048;
    float ei = 0.f, ef = 0.f, eg = 0.f, eo = 0.f;
    if (embW) {
        const float* er = embW + (long)y[b * T + t] * 2048;
        ei = er[d]; ef = er[512 + d]; eg = er[1024 + d]; eo = er[1536 + d];
    }
    const float gi = gr[d]        + bih[d]        + bhh[d]        + ei;
    const float gf = gr[512 + d]  + bih[512 + d]  + bhh[512 + d]  + ef;
    const float gg = gr[1024 + d] + bih[1024 + d] + bhh[1024 + d] + eg;
    const float go = gr[1536 + d] + bih[1536 + d] + bhh[1536 + d] + eo;
    const float c = sigmoidf_(gf) * cst[g] + sigmoidf_(gi) * tanhf(gg);
    cst[g] = c;
    const float h = sigmoidf_(go) * tanhf(c);
    const long pd = frag_perm(d);
    outA[(long)b * strideA + pd] = (_Float16)h;
    if (outB) outB[(long)b * strideB + pd] = (_Float16)h;
}

// ---------------------------------------------------------------------------
// Host side
// ---------------------------------------------------------------------------
static inline dim3 gemm_grid(int M, int N) { return dim3((N + 127) / 128, M / 32, 1); }

extern "C" void kernel_launch(void* const* d_in, const int* in_sizes, int n_in,
                              void* d_out, int out_size, void* d_ws, size_t ws_size,
                              hipStream_t stream) {
    // Input order: jax tree flatten (dict keys sorted) of setup_inputs():
    // h, y, params{lstm0{Whh,Wih,bhh,bih}, lstm1{...}, out{W0,W1,Wf,b0,b1,be0,be1,bef,bf,g0,g1,gf},
    //              phi{W0,Wf,b0,be0,bef,bf,g0,gf}, psi{...}}, sample_prob
    const float* h     = (const float*)d_in[0];
    const int*   y     = (const int*)  d_in[1];
    const float* l0Whh = (const float*)d_in[2];
    const float* l0Wih = (const float*)d_in[3];   // [10512, 2048]
    const float* l0bhh = (const float*)d_in[4];
    const float* l0bih = (const float*)d_in[5];
    const float* l1Whh = (const float*)d_in[6];
    const float* l1Wih = (const float*)d_in[7];
    const float* l1bhh = (const float*)d_in[8];
    const float* l1bih = (const float*)d_in[9];
    const float* oW0   = (const float*)d_in[10];
    const float* oW1   = (const float*)d_in[11];
    const float* oWf   = (const float*)d_in[12];  // [512, 10000]
    const float* ob0   = (const float*)d_in[13];
    const float* ob1   = (const float*)d_in[14];
    const float* obe0  = (const float*)d_in[15];
    const float* obe1  = (const float*)d_in[16];
    const float* obef  = (const float*)d_in[17];
    const float* obf   = (const float*)d_in[18];
    const float* og0   = (const float*)d_in[19];
    const float* og1   = (const float*)d_in[20];
    const float* ogf   = (const float*)d_in[21];
    const float* pW0   = (const float*)d_in[22];  // phi
    const float* pWf   = (const float*)d_in[23];
    const float* pb0   = (const float*)d_in[24];
    const float* pbe0  = (const float*)d_in[25];
    const float* pbef  = (const float*)d_in[26];
    const float* pbf   = (const float*)d_in[27];
    const float* pg0   = (const float*)d_in[28];
    const float* pgf   = (const float*)d_in[29];
    const float* qW0   = (const float*)d_in[30];  // psi
    const float* qWf   = (const float*)d_in[31];
    const float* qb0   = (const float*)d_in[32];
    const float* qbe0  = (const float*)d_in[33];
    const float* qbef  = (const float*)d_in[34];
    const float* qbf   = (const float*)d_in[35];
    const float* qg0   = (const float*)d_in[36];
    const float* qgf   = (const float*)d_in[37];
    // d_in[38] = sample_prob == 0.0 -> teacher forcing (sampling path is dead)

    float* out = (float*)d_out;  // [B, T, V]

    // ---- workspace bump allocator (~105 MB required) ----
    char* base = (char*)d_ws;
    size_t off = 0;
    auto alloc = [&](size_t bytes) -> void* {
        void* p = base + off;
        off = (off + bytes + 255) & ~(size_t)255;
        return p;
    };
    const long HR = (long)BATCH * RLEN;        // 32768 rows
    _Float16* phiW0T = (_Float16*)alloc(512 * 512 * 2);
    _Float16* phiWfT = (_Float16*)alloc(512 * 512 * 2);
    _Float16* psiW0T = (_Float16*)alloc(512 * 512 * 2);
    _Float16* psiWfT = (_Float16*)alloc(512 * 512 * 2);
    _Float16* oW0T   = (_Float16*)alloc(512 * 512 * 2);
    _Float16* oW1T   = (_Float16*)alloc(512 * 512 * 2);
    _Float16* oWfT   = (_Float16*)alloc((size_t)VOCAB * 512 * 2);
    _Float16* l0WT   = (_Float16*)alloc((size_t)2048 * 1024 * 2);
    _Float16* l1WT   = (_Float16*)alloc((size_t)2048 * 1024 * 2);
    _Float16* hA16   = (_Float16*)alloc((size_t)HR * HID * 2);   // h f16, reused for LN stage out
    _Float16* hpsi16 = (_Float16*)alloc((size_t)HR * HID * 2);
    const int CH = 4096;                                          // psi precompute row chunk
    float*    big32  = (float*)alloc((size_t)CH * HID * 4);
    float*    e32    = (float*)alloc((size_t)BATCH * RLEN * 4);
    float*    alpha  = (float*)alloc((size_t)BATCH * RLEN * 4);
    float*    g32    = (float*)alloc((size_t)BATCH * 2048 * 4);
    float*    gbig   = (float*)alloc((size_t)BATCH * VOCAB * 4);
    _Float16* xA0    = (_Float16*)alloc((size_t)BATCH * 1024 * 2); // [c | s0]
    _Float16* xA1    = (_Float16*)alloc((size_t)BATCH * 1024 * 2); // [s0 | s1]
    _Float16* s1_16  = (_Float16*)alloc((size_t)BATCH * HID * 2);
    _Float16* sphi16 = (_Float16*)alloc((size_t)BATCH * HID * 2);
    _Float16* tphi16 = (_Float16*)alloc((size_t)BATCH * HID * 2);
    _Float16* t1_16  = (_Float16*)alloc((size_t)BATCH * HID * 2);
    _Float16* t2_16  = (_Float16*)alloc((size_t)BATCH * HID * 2);
    float*    t1_32  = (float*)alloc((size_t)BATCH * HID * 4);
    float*    c0     = (float*)alloc((size_t)BATCH * HID * 4);
    float*    c1     = (float*)alloc((size_t)BATCH * HID * 4);
    (void)ws_size; (void)in_sizes; (void)n_in; (void)out_size;

    const dim3 blk(256);
    auto tcvt = [&](const float* s, _Float16* d, int K, int N, int dstStride, int dstOff) {
        long n = (long)K * N;
        transpose_cvt_kernel<<<dim3((unsigned)((n + 255) / 256)), blk, 0, stream>>>(s, d, K, N, dstStride, dstOff);
    };

    // ---- one-time: weight convert/transpose (into permuted-K layout) ----
    tcvt(pW0, phiW0T, 512, 512, 512, 0);
    tcvt(pWf, phiWfT, 512, 512, 512, 0);
    tcvt(qW0, psiW0T, 512, 512, 512, 0);
    tcvt(qWf, psiWfT, 512, 512, 512, 0);
    tcvt(oW0, oW0T,   512, 512, 512, 0);
    tcvt(oW1, oW1T,   512, 512, 512, 0);
    tcvt(oWf, oWfT,   512, VOCAB, 512, 0);
    tcvt(l0Wih + (long)VOCAB * 2048, l0WT, 512, 2048, 1024, 0);    // c-part of Wih
    tcvt(l0Whh,                       l0WT, 512, 2048, 1024, 512); // Whh
    tcvt(l1Wih, l1WT, 512, 2048, 1024, 0);
    tcvt(l1Whh, l1WT, 512, 2048, 1024, 512);

    cvt_f16_perm_kernel<<<dim3(4096), blk, 0, stream>>>(h, hA16, HR * HID);

    // ---- zero initial decoder state ----
    hipMemsetAsync(xA0,   0, (size_t)BATCH * 1024 * 2, stream);
    hipMemsetAsync(xA1,   0, (size_t)BATCH * 1024 * 2, stream);
    hipMemsetAsync(s1_16, 0, (size_t)BATCH * HID * 2, stream);
    hipMemsetAsync(c0,    0, (size_t)BATCH * HID * 4, stream);
    hipMemsetAsync(c1,    0, (size_t)BATCH * HID * 4, stream);

    // ---- precompute h_psi = mlp2(psi, h), chunked over rows ----
    for (long cr = 0; cr < HR; cr += CH) {
        gemm_f16_wmma_kernel<<<gemm_grid(CH, 512), blk, 0, stream>>>(
            hA16 + cr * HID, HID, psiW0T, 512, big32, 512, 512, 512);
        bias_act_ln_kernel<<<dim3(CH), blk, 0, stream>>>(
            big32, qb0, nullptr, qg0, qbe0, hA16 + cr * HID, HID, nullptr, 0, HID, 1);
        gemm_f16_wmma_kernel<<<gemm_grid(CH, 512), blk, 0, stream>>>(
            hA16 + cr * HID, HID, psiWfT, 512, big32, 512, 512, 512);
        bias_act_ln_kernel<<<dim3(CH), blk, 0, stream>>>(
            big32, qbf, nullptr, qgf, qbef, hpsi16 + cr * HID, HID, nullptr, 0, HID, 0);
    }

    // ---- decode loop (strictly sequential dependency chain) ----
    for (int t = 0; t < TLEN; ++t) {
        // s_phi = mlp2(phi, s1_prev)
        gemm_f16_wmma_kernel<<<gemm_grid(32, 512), blk, 0, stream>>>(
            s1_16, HID, phiW0T, 512, g32, 512, 512, 512);
        bias_act_ln_kernel<<<dim3(32), blk, 0, stream>>>(
            g32, pb0, nullptr, pg0, pbe0, tphi16, HID, nullptr, 0, HID, 1);
        gemm_f16_wmma_kernel<<<gemm_grid(32, 512), blk, 0, stream>>>(
            tphi16, HID, phiWfT, 512, g32, 512, 512, 512);
        bias_act_ln_kernel<<<dim3(32), blk, 0, stream>>>(
            g32, pbf, nullptr, pgf, pbef, sphi16, HID, nullptr, 0, HID, 0);

        // attention: e -> softmax -> c (c written into xA0[:, 0:512])
        attn_e_kernel<<<dim3(BATCH * RLEN / 8), blk, 0, stream>>>(sphi16, hpsi16, e32);
        softmax_rows_kernel<<<dim3(BATCH), blk, 0, stream>>>(e32, alpha, RLEN);
        attn_c_kernel<<<dim3(BATCH), blk, 0, stream>>>(alpha, hpsi16, xA0);

        // lstm0: gates = [c|s0] @ [Wih_c;Whh] ; one-hot part via gathered Wih row
        gemm_f16_wmma_kernel<<<gemm_grid(32, 2048), blk, 0, stream>>>(
            xA0, 1024, l0WT, 1024, g32, 2048, 2048, 1024);
        lstm_cell_kernel<<<dim3(64), blk, 0, stream>>>(
            g32, l0bih, l0bhh, l0Wih, y, t, TLEN, c0, xA0 + 512, 1024, xA1, 1024);

        // lstm1: gates = [s0|s1] @ [Wih;Whh]
        gemm_f16_wmma_kernel<<<gemm_grid(32, 2048), blk, 0, stream>>>(
            xA1, 1024, l1WT, 1024, g32, 2048, 2048, 1024);
        lstm_cell_kernel<<<dim3(64), blk, 0, stream>>>(
            g32, l1bih, l1bhh, nullptr, nullptr, 0, 0, c1, xA1 + 512, 1024, s1_16, HID);

        // out mlp3
        gemm_f16_wmma_kernel<<<gemm_grid(32, 512), blk, 0, stream>>>(
            s1_16, HID, oW0T, 512, g32, 512, 512, 512);
        bias_act_ln_kernel<<<dim3(32), blk, 0, stream>>>(
            g32, ob0, nullptr, og0, obe0, t1_16, HID, t1_32, HID, HID, 1);
        gemm_f16_wmma_kernel<<<gemm_grid(32, 512), blk, 0, stream>>>(
            t1_16, HID, oW1T, 512, g32, 512, 512, 512);
        bias_act_ln_kernel<<<dim3(32), blk, 0, stream>>>(
            g32, ob1, t1_32, og1, obe1, t2_16, HID, nullptr, 0, HID, 1);
        gemm_f16_wmma_kernel<<<gemm_grid(32, VOCAB), blk, 0, stream>>>(
            t2_16, HID, oWfT, 512, gbig, VOCAB, VOCAB, 512);
        bias_act_ln_kernel<<<dim3(32), blk, 0, stream>>>(
            gbig, obf, nullptr, ogf, obef, nullptr, 0,
            out + (long)t * VOCAB, (long)TLEN * VOCAB, VOCAB, 0);
    }
}
